// LTU_15350213116178
// MI455X (gfx1250) — compile-verified, matrix operand order checked by reading, not run on hardware
//
#include <hip/hip_runtime.h>

typedef __attribute__((ext_vector_type(8))) int          v8i;
typedef __attribute__((ext_vector_type(4))) unsigned int v4u;

// ---------------------------------------------------------------------------
// Kernel 1: zero the per-row negative counters in workspace (must run every
// call: the count kernel accumulates with atomics and the harness does not
// re-initialize d_ws between replays).
// ---------------------------------------------------------------------------
__global__ void ltu_zero_counts(int* __restrict__ cnt, int n) {
    int i = blockIdx.x * blockDim.x + threadIdx.x;
    if (i < n) cnt[i] = 0;
}

// ---------------------------------------------------------------------------
// Kernel 2: per-row negative count via V_WMMA_I32_16X16X64_IU8.
//
//   A[m,k] = 0x80 if W[row_m, k] < 0 else 0   (u8; sign byte via v_perm_b32)
//   B[k,n] = 1                                 (u8, layout-invariant)
//   D[m,n] = 128 * neg_count (replicated across n), accumulated i32 over K,
//   published as (D >> 7).
//
// Block = 16 rows. 8 waves per block, each wave owns itersPerWave*64 columns.
// 8-bit A 16x64 layout: lanes 0-15 hold rows, lanes 16-31 the same rows with
// K shifted by +8; VGPR j covers K-offset {0,4,16,20,32,36,48,52}[j] (+hi*8).
// All float4 loads are 16B aligned -> global_load_b128, marked non-temporal
// (single-pass 256MB stream > 192MB L2).
//
// Sign-byte gather per float4 (3x v_perm_b32 + 1x v_and, selectors 0-7 only):
//   t0 = perm(w1, w0, 0x0703)      -> {w0.b3, w1.b3, x, x}
//   t1 = perm(w3, w2, 0x07030000)  -> {x, x, w2.b3, w3.b3}
//   t  = perm(t1, t0, 0x07060100)  -> {w0.b3, w1.b3, w2.b3, w3.b3}
//   a  = t & 0x80808080            -> bytes in {0, 0x80}
// ---------------------------------------------------------------------------
__global__ void ltu_neg_count_wmma(const float* __restrict__ W,
                                   int* __restrict__ cnt,
                                   int nCols, int itersPerWave) {
    const int tid    = threadIdx.x;
    const int lane   = tid & 31;          // wave32
    const int waveId = tid >> 5;
    const int m      = lane & 15;         // row within the 16-row block
    const int hi     = lane >> 4;         // half-wave K-offset selector
    const int row    = blockIdx.x * 16 + m;

    const float* rowPtr = W + (size_t)row * (size_t)nCols;
    const int k0 = waveId * (itersPerWave * 64) + hi * 8;

    v8i b;
#pragma unroll
    for (int j = 0; j < 8; ++j) b[j] = 0x01010101;   // all-ones u8 B matrix

    v8i c = {};                                      // i32 accumulator (C/D)

    for (int it = 0; it < itersPerWave; ++it) {
        const int k = k0 + it * 64;
        v8i a;
#pragma unroll
        for (int j = 0; j < 8; ++j) {
            const int off = (j >> 1) * 16 + (j & 1) * 4;   // 0,4,16,20,32,36,48,52
            v4u w = __builtin_nontemporal_load(
                        (const v4u*)(rowPtr + k + off));   // b128 NT load
            unsigned t0 = __builtin_amdgcn_perm(w[1], w[0], 0x00000703u);
            unsigned t1 = __builtin_amdgcn_perm(w[3], w[2], 0x07030000u);
            unsigned t  = __builtin_amdgcn_perm(t1, t0, 0x07060100u);
            a[j] = (int)(t & 0x80808080u);   // 0x80 per negative element
        }
        // 7 args: (sgn_a, A, sgn_b, B, C, reuse_a, reuse_b) -> unsigned u8
        c = __builtin_amdgcn_wmma_i32_16x16x64_iu8(false, a, false, b, c,
                                                   false, false);
    }

    // i32 16x16 C/D layout: VGPR r, lanes 0-15 -> M=r ; lanes 16-31 -> M=r+8.
    // Every N column holds the same row-sum; lane 0 publishes rows 0-7,
    // lane 16 publishes rows 8-15.  Counts are scaled by 128 -> shift out.
    if (m == 0) {
        const int rbase = blockIdx.x * 16 + hi * 8;
#pragma unroll
        for (int r = 0; r < 8; ++r)
            atomicAdd(&cnt[rbase + r], c[r] >> 7);
    }
}

// ---------------------------------------------------------------------------
// Kernel 3: threshold.  tau0 = float(nCols) * 0.6f, matching the f32 math in
// the reference (8192 is a power of two, so the product is exact in f32).
// ---------------------------------------------------------------------------
__global__ void ltu_finalize(const float* __restrict__ x,
                             const int* __restrict__ cnt,
                             float* __restrict__ out, int n, float tau0) {
    int i = blockIdx.x * blockDim.x + threadIdx.x;
    if (i < n) {
        float tau = tau0 - (float)cnt[i];
        out[i] = (x[i] < tau) ? 0.0f : 1.0f;
    }
}

extern "C" void kernel_launch(void* const* d_in, const int* in_sizes, int n_in,
                              void* d_out, int out_size, void* d_ws, size_t ws_size,
                              hipStream_t stream) {
    const float* x = (const float*)d_in[0];   // [nRows]
    const float* W = (const float*)d_in[1];   // [nRows, nCols]
    float* out     = (float*)d_out;
    int*   cnt     = (int*)d_ws;              // nRows i32 counters (32 KB)

    const int nRows = in_sizes[0];                            // 8192
    const int nCols = (int)((long long)in_sizes[1] / nRows);  // 8192

    // 8 waves per block, each wave covers nCols/8 columns in 64-wide steps.
    const int itersPerWave = nCols / (8 * 64);                // 16
    const float tau0 = (float)nCols * 0.6f;

    ltu_zero_counts<<<(nRows + 255) / 256, 256, 0, stream>>>(cnt, nRows);

    ltu_neg_count_wmma<<<nRows / 16, 256, 0, stream>>>(W, cnt, nCols,
                                                       itersPerWave);

    ltu_finalize<<<(nRows + 255) / 256, 256, 0, stream>>>(x, cnt, out,
                                                          nRows, tau0);
}
